// MultiheadAttention_50087908606210
// MI455X (gfx1250) — matrix operand order repaired and checked by reading, not compiled, not run on hardware
//
#include <hip/hip_runtime.h>

typedef __attribute__((ext_vector_type(16))) __bf16 v16bf;
typedef __attribute__((ext_vector_type(8)))  __bf16 v8bf;
typedef __attribute__((ext_vector_type(4)))  __bf16 v4bf;
typedef __attribute__((ext_vector_type(8)))  float  v8f;

#define BDIM 2
#define TDIM 2048
#define CDIM 1024
#define HDIM 16
#define DDIM 64
#define MROWS (BDIM * TDIM)      // 4096 tokens
#define NQKV (3 * CDIM)          // 3072
#define KCHUNKS (CDIM / 32)      // 32 (even)

static __device__ __forceinline__ v8f zero8() {
  v8f z = {0.f, 0.f, 0.f, 0.f, 0.f, 0.f, 0.f, 0.f};
  return z;
}

// Build a 16-element bf16 fragment from two contiguous 16-byte loads.
static __device__ __forceinline__ v16bf frag16(const __bf16* p0, const __bf16* p1) {
  v8bf lo = *(const v8bf*)p0;
  v8bf hi = *(const v8bf*)p1;
  v16bf r;
#pragma unroll
  for (int i = 0; i < 8; ++i) { r[i] = lo[i]; r[i + 8] = hi[i]; }
  return r;
}

static __device__ __forceinline__ v8f wmma_bf16(v16bf a, v16bf b, v8f c) {
  return __builtin_amdgcn_wmma_f32_16x16x32_bf16(false, a, false, b, (short)0, c,
                                                 false, false);
}

// Async global->LDS copy of 16 bytes per lane (ASYNCcnt path, no VGPR data).
// LDS operand = workgroup-relative byte offset = low 32 bits of generic pointer.
static __device__ __forceinline__ void async_copy_b128(void* lds_dst,
                                                       const void* gsrc) {
  unsigned int l = (unsigned int)(unsigned long long)lds_dst;
  asm volatile("global_load_async_to_lds_b128 %0, %1, off"
               :: "v"(l), "v"(gsrc) : "memory");
}

static __device__ __forceinline__ void wait_async0() {
  asm volatile("s_wait_asynccnt 0" ::: "memory");
}

// ---------------------------------------------------------------- convert
__global__ void cvt_f32_bf16_kernel(const float4* __restrict__ s,
                                    v4bf* __restrict__ d, int n4) {
  int i = blockIdx.x * blockDim.x + threadIdx.x;
  int stride = gridDim.x * blockDim.x;
  for (; i < n4; i += stride) {
    float4 f = s[i];
    v4bf o = {(__bf16)f.x, (__bf16)f.y, (__bf16)f.z, (__bf16)f.w};
    d[i] = o;
  }
}

// ---------------------------------------------------------------- GEMM core
// Block: 256 thr = 8 waves. Block tile: 128 rows x 128 cols.
// Wave w: rows [row0, row0+16); all 8 n-subtiles of 16 cols.
// B tile (128 cols x 32 k, bf16, 8KB) staged via async copies, double-buffered,
// k-loop explicitly unrolled 2x over the two buffers (no register rotation).

static __device__ __forceinline__ void copy_b_async(const __bf16* __restrict__ Bm,
                                                    __bf16* dst, int n0, int k0) {
  const int tid = threadIdx.x;
#pragma unroll
  for (int j = 0; j < 2; ++j) {
    int c = tid + j * 256;   // 512 chunks of 16B
    int col = c >> 2;        // 0..127
    int sub = c & 3;         // 0..3
    async_copy_b128(dst + col * 32 + sub * 8,
                    Bm + (size_t)(n0 + col) * CDIM + k0 + sub * 8);
  }
}

static __device__ __forceinline__ void gemm_compute_chunk(const __bf16* bcur,
                                                          v16bf a, v8f* acc,
                                                          int l16, int half) {
  v16bf vf[8];
#pragma unroll
  for (int t = 0; t < 8; ++t) {
    const __bf16* bp = bcur + (t * 16 + l16) * 32 + half * 8;
    vf[t] = frag16(bp, bp + 16);
  }
#pragma unroll
  for (int t = 0; t < 8; ++t) acc[t] = wmma_bf16(a, vf[t], acc[t]);
}

static __device__ __forceinline__ void gemm_core(const __bf16* __restrict__ Am,
                                                 const __bf16* __restrict__ Bm,
                                                 __bf16* bsh,  // 2 * 128*32
                                                 int row0, int n0, v8f* acc) {
  const int lane = threadIdx.x & 31;
  const int half = lane >> 4;
  const int l16  = lane & 15;
#pragma unroll
  for (int t = 0; t < 8; ++t) acc[t] = zero8();

  const __bf16* arow = Am + (size_t)(row0 + l16) * CDIM + half * 8;
  __bf16* bsh0 = bsh;
  __bf16* bsh1 = bsh + 128 * 32;

  copy_b_async(Bm, bsh0, n0, 0);
  v16bf a0 = frag16(arow, arow + 16);
  wait_async0();
  __syncthreads();

  for (int kc = 0; kc < KCHUNKS; kc += 2) {
    // ---- stage A: compute chunk kc from bsh0; prefetch kc+1 -> bsh1
    const __bf16* ap1 = arow + (kc + 1) * 32;
    v16bf a1 = frag16(ap1, ap1 + 16);
    copy_b_async(Bm, bsh1, n0, (kc + 1) * 32);
    gemm_compute_chunk(bsh0, a0, acc, l16, half);
    wait_async0();
    __syncthreads();

    // ---- stage B: compute chunk kc+1 from bsh1; prefetch kc+2 -> bsh0
    if (kc + 2 < KCHUNKS) {
      const __bf16* ap2 = arow + (kc + 2) * 32;
      a0 = frag16(ap2, ap2 + 16);
      copy_b_async(Bm, bsh0, n0, (kc + 2) * 32);
    }
    gemm_compute_chunk(bsh1, a1, acc, l16, half);
    wait_async0();
    __syncthreads();
  }
}

// ---------------------------------------------------------------- QKV GEMM
__global__ __launch_bounds__(256) void qkv_gemm_kernel(
    const __bf16* __restrict__ xb, const __bf16* __restrict__ wab,
    __bf16* __restrict__ qb, __bf16* __restrict__ kb, __bf16* __restrict__ vtb) {
  __shared__ __align__(16) __bf16 bsh[2 * 128 * 32];
  const int wave = threadIdx.x >> 5;
  const int lane = threadIdx.x & 31;
  const int half = lane >> 4;
  const int l16  = lane & 15;
  const int row0 = blockIdx.y * 128 + wave * 16;
  const int n0   = blockIdx.x * 128;

  v8f acc[8];
  gemm_core(xb, wab, bsh, row0, n0, acc);

#pragma unroll
  for (int t = 0; t < 8; ++t) {
#pragma unroll
    for (int r = 0; r < 8; ++r) {
      int row  = row0 + r + half * 8;           // global token
      int bcol = n0 + t * 16 + l16;             // 0..3071
      int which = bcol >> 10;                   // 0=q 1=k 2=v
      int rem   = bcol & 1023;
      int h = rem >> 6, d = rem & 63;
      int bb = row >> 11, tt = row & (TDIM - 1);
      size_t bh = (size_t)bb * HDIM + h;
      __bf16 bv = (__bf16)acc[t][r];
      if (which == 0)      qb [(bh * TDIM + tt) * DDIM + d] = bv;
      else if (which == 1) kb [(bh * TDIM + tt) * DDIM + d] = bv;
      else                 vtb[(bh * DDIM + d) * TDIM + tt] = bv;
    }
  }
}

// ---------------------------------------------------------------- proj GEMM
__global__ __launch_bounds__(256) void proj_gemm_kernel(
    const __bf16* __restrict__ yb, const __bf16* __restrict__ wpb,
    float* __restrict__ out) {
  __shared__ __align__(16) __bf16 bsh[2 * 128 * 32];
  const int wave = threadIdx.x >> 5;
  const int lane = threadIdx.x & 31;
  const int half = lane >> 4;
  const int l16  = lane & 15;
  const int row0 = blockIdx.y * 128 + wave * 16;
  const int n0   = blockIdx.x * 128;

  v8f acc[8];
  gemm_core(yb, wpb, bsh, row0, n0, acc);

#pragma unroll
  for (int t = 0; t < 8; ++t)
#pragma unroll
    for (int r = 0; r < 8; ++r)
      out[(size_t)(row0 + r + half * 8) * CDIM + n0 + t * 16 + l16] = acc[t][r];
}

// ---------------------------------------------------------------- flash attention
// Grid: (T/128, B*H). Block: 8 waves. K (32x64) and Vt (64x32) tiles staged in
// double-buffered LDS via async copies, shared by all waves. Each wave: online
// softmax over its 16 q-rows; S = Q K^T and Y += P V via WMMA.

static __device__ __forceinline__ void copy_kv_async(const __bf16* __restrict__ K,
                                                     const __bf16* __restrict__ Vt,
                                                     __bf16* kdst, __bf16* vdst,
                                                     int kcol0) {
  const int tid = threadIdx.x;
  {  // K tile: 32 rows x 64 d
    int row = tid >> 3, sub = tid & 7;
    async_copy_b128(kdst + row * 64 + sub * 8,
                    K + (size_t)(kcol0 + row) * DDIM + sub * 8);
  }
  {  // Vt tile: 64 d-rows x 32 k
    int row = tid >> 2, sub = tid & 3;
    async_copy_b128(vdst + row * 32 + sub * 8,
                    Vt + (size_t)row * TDIM + kcol0 + sub * 8);
  }
}

__global__ __launch_bounds__(256) void flash_attn_kernel(
    const __bf16* __restrict__ qb, const __bf16* __restrict__ kb,
    const __bf16* __restrict__ vtb, __bf16* __restrict__ yb) {
  const int wave = threadIdx.x >> 5;
  const int lane = threadIdx.x & 31;
  const int half = lane >> 4;
  const int l16  = lane & 15;
  const int bh   = blockIdx.y;
  const int qrow0_blk = blockIdx.x * 128;
  const int qrow0 = qrow0_blk + wave * 16;
  const float SCALE = 0.125f;                   // 1/sqrt(64)

  const __bf16* Q  = qb  + (size_t)bh * TDIM * DDIM;
  const __bf16* K  = kb  + (size_t)bh * TDIM * DDIM;
  const __bf16* Vt = vtb + (size_t)bh * DDIM * TDIM;

  __shared__ __align__(16) __bf16 ksh[2 * 32 * 64];   // 8KB
  __shared__ __align__(16) __bf16 vsh[2 * 64 * 32];   // 8KB
  __shared__ __align__(16) __bf16 psh[8][16 * 32];    // 8KB (per-wave P tiles)
  __bf16* pw = &psh[wave][0];

  // Q fragments for d=0..31 and d=32..63, held for the whole loop
  const __bf16* qrow = Q + (size_t)(qrow0 + l16) * DDIM;
  v16bf qa0 = frag16(qrow + half * 8,      qrow + half * 8 + 16);
  v16bf qa1 = frag16(qrow + 32 + half * 8, qrow + 32 + half * 8 + 16);

  v8f Y[4];
#pragma unroll
  for (int t = 0; t < 4; ++t) Y[t] = zero8();
  float mrow[8], lrow[8];
#pragma unroll
  for (int r = 0; r < 8; ++r) { mrow[r] = -1e30f; lrow[r] = 0.f; }

  const int kbmax     = (qrow0 + 15) >> 5;       // this wave's causal limit
  const int kbmax_blk = (qrow0_blk + 127) >> 5;  // block's causal limit

  copy_kv_async(K, Vt, ksh, vsh, 0);
  wait_async0();
  __syncthreads();

  for (int kblk = 0; kblk <= kbmax_blk; ++kblk) {
    const int cur = kblk & 1;
    if (kblk + 1 <= kbmax_blk)
      copy_kv_async(K, Vt, ksh + ((kblk + 1) & 1) * (32 * 64),
                    vsh + ((kblk + 1) & 1) * (64 * 32), (kblk + 1) * 32);

    if (kblk <= kbmax) {
      const int kcol0 = kblk * 32;
      const __bf16* kbase = ksh + cur * (32 * 64);
      const __bf16* vbase = vsh + cur * (64 * 32);

      // batch-load all four K fragments, then 4 WMMA (partial DS waits)
      const __bf16* kr0 = kbase + l16 * 64 + half * 8;
      const __bf16* kr1 = kbase + (16 + l16) * 64 + half * 8;
      v16bf kf00 = frag16(kr0,      kr0 + 16);   // tile0, d 0..31
      v16bf kf01 = frag16(kr0 + 32, kr0 + 48);   // tile0, d 32..63
      v16bf kf10 = frag16(kr1,      kr1 + 16);   // tile1, d 0..31
      v16bf kf11 = frag16(kr1 + 32, kr1 + 48);   // tile1, d 32..63

      v8f s[2];
      s[0] = zero8(); s[1] = zero8();
      s[0] = wmma_bf16(qa0, kf00, s[0]);
      s[0] = wmma_bf16(qa1, kf01, s[0]);
      s[1] = wmma_bf16(qa0, kf10, s[1]);
      s[1] = wmma_bf16(qa1, kf11, s[1]);

      // scale + causal mask
#pragma unroll
      for (int tile = 0; tile < 2; ++tile) {
#pragma unroll
        for (int r = 0; r < 8; ++r) {
          float v = s[tile][r] * SCALE;
          int col = kcol0 + tile * 16 + l16;
          int row = qrow0 + r + half * 8;
          if (col > row) v = -1e30f;
          s[tile][r] = v;
        }
      }

      // row max across the 16 lanes of this half-group
      float rmax[8];
#pragma unroll
      for (int r = 0; r < 8; ++r) rmax[r] = fmaxf(s[0][r], s[1][r]);
#pragma unroll
      for (int off = 1; off < 16; off <<= 1)
#pragma unroll
        for (int r = 0; r < 8; ++r)
          rmax[r] = fmaxf(rmax[r], __shfl_xor(rmax[r], off, 32));

      float corr[8], rsum[8];
#pragma unroll
      for (int r = 0; r < 8; ++r) {
        float mn = fmaxf(mrow[r], rmax[r]);
        corr[r] = __expf(mrow[r] - mn);
        mrow[r] = mn;
        float p0 = __expf(s[0][r] - mn);
        float p1 = __expf(s[1][r] - mn);
        s[0][r] = p0; s[1][r] = p1;
        rsum[r] = p0 + p1;
      }
#pragma unroll
      for (int off = 1; off < 16; off <<= 1)
#pragma unroll
        for (int r = 0; r < 8; ++r) rsum[r] += __shfl_xor(rsum[r], off, 32);
#pragma unroll
      for (int r = 0; r < 8; ++r) lrow[r] = lrow[r] * corr[r] + rsum[r];
#pragma unroll
      for (int t = 0; t < 4; ++t)
#pragma unroll
        for (int r = 0; r < 8; ++r) Y[t][r] *= corr[r];

      // P (f32, C/D layout) -> bf16 -> per-wave LDS tile [16 x 32] row-major
#pragma unroll
      for (int tile = 0; tile < 2; ++tile)
#pragma unroll
        for (int r = 0; r < 8; ++r)
          pw[(r + half * 8) * 32 + tile * 16 + l16] = (__bf16)s[tile][r];
      __asm__ volatile("s_wait_dscnt 0" ::: "memory");

      // reload P as A-fragment; batch-load all four V fragments, then 4 WMMA
      const __bf16* prow = pw + l16 * 32 + half * 8;
      v16bf pa = frag16(prow, prow + 16);
      v16bf vf[4];
#pragma unroll
      for (int t = 0; t < 4; ++t) {
        const __bf16* vr = vbase + (t * 16 + l16) * 32 + half * 8;
        vf[t] = frag16(vr, vr + 16);
      }
#pragma unroll
      for (int t = 0; t < 4; ++t) Y[t] = wmma_bf16(pa, vf[t], Y[t]);
    }
    wait_async0();
    __syncthreads();
  }

  // epilogue: y = Y / l, scatter to [B,T,C] bf16
  const int b = bh / HDIM, h = bh % HDIM;
#pragma unroll
  for (int t = 0; t < 4; ++t) {
#pragma unroll
    for (int r = 0; r < 8; ++r) {
      float yv = Y[t][r] / lrow[r];
      size_t token = (size_t)b * TDIM + qrow0 + r + half * 8;
      yb[token * CDIM + h * DDIM + t * 16 + l16] = (__bf16)yv;
    }
  }
}

// ---------------------------------------------------------------- launch
extern "C" void kernel_launch(void* const* d_in, const int* in_sizes, int n_in,
                              void* d_out, int out_size, void* d_ws, size_t ws_size,
                              hipStream_t stream) {
  const float* x      = (const float*)d_in[0];
  const float* w_attn = (const float*)d_in[1];
  const float* w_proj = (const float*)d_in[2];
  // d_in[3] = causal mask, unused (computed analytically)
  float* out = (float*)d_out;

  const size_t nX  = (size_t)BDIM * TDIM * CDIM;   // 4,194,304
  const size_t nWA = (size_t)NQKV * CDIM;          // 3,145,728
  const size_t nWP = (size_t)CDIM * CDIM;          // 1,048,576

  __bf16* xb  = (__bf16*)d_ws;
  __bf16* wab = xb  + nX;
  __bf16* wpb = wab + nWA;
  __bf16* qb  = wpb + nWP;
  __bf16* kb  = qb  + nX;
  __bf16* vtb = kb  + nX;
  __bf16* yb  = vtb + nX;

  cvt_f32_bf16_kernel<<<2048, 256, 0, stream>>>((const float4*)x, (v4bf*)xb,
                                                (int)(nX / 4));
  cvt_f32_bf16_kernel<<<2048, 256, 0, stream>>>((const float4*)w_attn, (v4bf*)wab,
                                                (int)(nWA / 4));
  cvt_f32_bf16_kernel<<<1024, 256, 0, stream>>>((const float4*)w_proj, (v4bf*)wpb,
                                                (int)(nWP / 4));

  qkv_gemm_kernel<<<dim3(NQKV / 128, MROWS / 128), 256, 0, stream>>>(xb, wab, qb,
                                                                     kb, vtb);
  flash_attn_kernel<<<dim3(TDIM / 128, BDIM * HDIM), 256, 0, stream>>>(qb, kb, vtb,
                                                                       yb);
  proj_gemm_kernel<<<dim3(CDIM / 128, MROWS / 128), 256, 0, stream>>>(yb, wpb, out);
}
